// MambaBlock_64630667870742
// MI455X (gfx1250) — matrix operand order repaired and checked by reading, not compiled, not run on hardware
//
#include <hip/hip_runtime.h>
#include <hip/hip_bf16.h>

#define LNUM 2
#define DMC  512
#define DIC  1024
#define DSC  16
#define DCV  4
#define DTRC 32
#define BBC  8
#define NNC  2048
#define BNC  (BBC * NNC)   /* 16384 rows */
#define EPSV 1e-5f

typedef __attribute__((ext_vector_type(16))) __bf16 v16bf;
typedef __attribute__((ext_vector_type(8)))  __bf16 v8bf;
typedef __attribute__((ext_vector_type(8)))  float  v8f;

// ---------------------------------------------------------------------------
// helpers
// ---------------------------------------------------------------------------
__device__ __forceinline__ float softplus_f(float x) {
    return (x > 20.f) ? x : log1pf(__expf(x));
}
__device__ __forceinline__ float silu_f(float x) {
    return x / (1.f + __expf(-x));
}

// Fragment k order used here (contiguous, same for A and B; 16-bit, 16x16x32):
//  A 16x32: lane m(0..15) holds K k0..k0+15;
//           lane m+16 holds k0+16..k0+31. Two 16B loads.
//  B 32x16: lane n(0..15) holds K k0..k0+15; lane n+16 holds k0+16..k0+31.
//           One contiguous 32B load (weights are [N,K] row-major).
__device__ __forceinline__ void load_frags(
    const __bf16* __restrict__ arow,
    const __bf16* __restrict__ w0, const __bf16* __restrict__ w1,
    const __bf16* __restrict__ w2, const __bf16* __restrict__ w3, int k0,
    v16bf& a, v16bf& b0, v16bf& b1, v16bf& b2, v16bf& b3) {
    union { v16bf v; v8bf h[2]; } au;
    au.h[0] = *(const v8bf*)(arow + k0);
    au.h[1] = *(const v8bf*)(arow + k0 + 8);
    a  = au.v;
    b0 = *(const v16bf*)(w0 + k0);
    b1 = *(const v16bf*)(w1 + k0);
    b2 = *(const v16bf*)(w2 + k0);
    b3 = *(const v16bf*)(w3 + k0);
}

#define WMMA4(a_, b0_, b1_, b2_, b3_)                                                        \
    acc0 = __builtin_amdgcn_wmma_f32_16x16x32_bf16(false, a_, false, b0_, (short)0, acc0,    \
                                                   false, false);                            \
    acc1 = __builtin_amdgcn_wmma_f32_16x16x32_bf16(false, a_, false, b1_, (short)0, acc1,    \
                                                   false, false);                            \
    acc2 = __builtin_amdgcn_wmma_f32_16x16x32_bf16(false, a_, false, b2_, (short)0, acc2,    \
                                                   false, false);                            \
    acc3 = __builtin_amdgcn_wmma_f32_16x16x32_bf16(false, a_, false, b3_, (short)0, acc3,    \
                                                   false, false)

// ---------------------------------------------------------------------------
// f32 -> bf16 conversion (4 elems/thread)
// ---------------------------------------------------------------------------
__global__ __launch_bounds__(256) void cvt_bf16_k(const float* __restrict__ s,
                                                  __bf16* __restrict__ d) {
    long i = ((long)blockIdx.x * blockDim.x + threadIdx.x) * 4;
#pragma unroll
    for (int j = 0; j < 4; ++j) d[i + j] = (__bf16)s[i + j];
}

// ---------------------------------------------------------------------------
// GEMM: C[M,Ncols] = A[M,K]_bf16 (row-major, lda) * W[Ncols,K]_bf16^T
// K is a compile-time constant -> branchless ping-pong pipeline, immediate
// load offsets, graduated waits. One wave -> 16x64 tile (4 accumulators,
// A reuse x4). Optional f32 / bf16 stores and fused bias+softplus.
// ---------------------------------------------------------------------------
template <int K, bool SOFTPLUS, bool WF32, bool WBF16>
__global__ __launch_bounds__(128) void gemm_wmma_bf16(
    const __bf16* __restrict__ A, int lda,
    const __bf16* __restrict__ W,
    const float* __restrict__ bias,
    float* __restrict__ C, __bf16* __restrict__ Cbf, int ldc, int ntilesN) {
    const int lane = threadIdx.x & 31;
    const int wv   = blockIdx.x * (blockDim.x >> 5) + (threadIdx.x >> 5);
    const int nb   = wv % ntilesN;
    const int mt   = wv / ntilesN;
    const int lo   = (lane < 16);
    const int mr   = lane & 15;

    v8f acc0 = {}, acc1 = {}, acc2 = {}, acc3 = {};

    const __bf16* arow = A + (long)(mt * 16 + mr) * lda + (lo ? 0 : 16);
    const __bf16* w0   = W + (long)(nb * 64 +  0 + mr) * K + (lo ? 0 : 16);
    const __bf16* w1   = W + (long)(nb * 64 + 16 + mr) * K + (lo ? 0 : 16);
    const __bf16* w2   = W + (long)(nb * 64 + 32 + mr) * K + (lo ? 0 : 16);
    const __bf16* w3   = W + (long)(nb * 64 + 48 + mr) * K + (lo ? 0 : 16);

    v16bf a0, b00, b01, b02, b03;   // phase 0
    v16bf a1, b10, b11, b12, b13;   // phase 1
    load_frags(arow, w0, w1, w2, w3, 0, a0, b00, b01, b02, b03);

#pragma unroll 4
    for (int k0 = 0; k0 < K; k0 += 64) {
        if (k0 + 32 < K)
            load_frags(arow, w0, w1, w2, w3, k0 + 32, a1, b10, b11, b12, b13);
        WMMA4(a0, b00, b01, b02, b03);
        if (k0 + 64 < K)
            load_frags(arow, w0, w1, w2, w3, k0 + 64, a0, b00, b01, b02, b03);
        if (k0 + 32 < K) {
            WMMA4(a1, b10, b11, b12, b13);
        }
    }

    const int rbase = mt * 16 + (lo ? 0 : 8);
    const int cbase = nb * 64 + mr;
#pragma unroll
    for (int r = 0; r < 8; ++r) {
        long  ro = (long)(rbase + r) * ldc + cbase;
        float v0 = acc0[r], v1 = acc1[r], v2 = acc2[r], v3 = acc3[r];
        if (SOFTPLUS) {
            v0 = softplus_f(v0 + bias[cbase +  0]);
            v1 = softplus_f(v1 + bias[cbase + 16]);
            v2 = softplus_f(v2 + bias[cbase + 32]);
            v3 = softplus_f(v3 + bias[cbase + 48]);
        }
        if (WF32) {
            C[ro +  0] = v0; C[ro + 16] = v1; C[ro + 32] = v2; C[ro + 48] = v3;
        }
        if (WBF16) {
            Cbf[ro +  0] = (__bf16)v0; Cbf[ro + 16] = (__bf16)v1;
            Cbf[ro + 32] = (__bf16)v2; Cbf[ro + 48] = (__bf16)v3;
        }
    }
}

// ---------------------------------------------------------------------------
// Causal depthwise conv (DC=4) + bias + SiLU. xz (bf16) = [xp | z], width 2*DI.
// Writes u in f32 (scan) and bf16 (x-projection A operand).
// ---------------------------------------------------------------------------
__global__ __launch_bounds__(256) void conv_silu_k(
    const __bf16* __restrict__ xz, const float* __restrict__ cw,
    const float* __restrict__ cb, float* __restrict__ u,
    __bf16* __restrict__ ubf) {
    int idx  = blockIdx.x * blockDim.x + threadIdx.x;   // BN*DI
    int d    = idx & (DIC - 1);
    int row  = idx >> 10;
    int n    = row & (NNC - 1);
    int rowb = row - n;                                  // b*NN
    float acc = cb[d];
#pragma unroll
    for (int j = 0; j < DCV; ++j) {
        int t = n - (DCV - 1) + j;
        if (t >= 0) acc += (float)xz[(long)(rowb + t) * (2 * DIC) + d] * cw[d * DCV + j];
    }
    float s = silu_f(acc);
    u[idx]   = s;
    ubf[idx] = (__bf16)s;
}

// ---------------------------------------------------------------------------
// Selective scan: one thread per (b,d), h[16] in registers, fused D-skip +
// SiLU(z) gating; emits bf16 directly for the out-projection.
// ---------------------------------------------------------------------------
__global__ __launch_bounds__(256) void scan_k(
    const float* __restrict__ dtf, const float* __restrict__ ubuf,
    const __bf16* __restrict__ xz, const float* __restrict__ dbc,
    const float* __restrict__ alog, const float* __restrict__ Dp,
    __bf16* __restrict__ yg) {
    int tid = blockIdx.x * blockDim.x + threadIdx.x;     // BB*DI
    int b   = tid >> 10;
    int d   = tid & (DIC - 1);

    float Areg[DSC];
#pragma unroll
    for (int s = 0; s < DSC; ++s) Areg[s] = -__expf(alog[d * DSC + s]);
    float h[DSC];
#pragma unroll
    for (int s = 0; s < DSC; ++s) h[s] = 0.f;
    const float Dd = Dp[d];

    for (int n = 0; n < NNC; ++n) {
        long row = (long)b * NNC + n;
        float dt = dtf[row * DIC + d];
        float uu = ubuf[row * DIC + d];
        float zz = (float)xz[row * (2 * DIC) + DIC + d];
        const float* Bp = dbc + row * 64 + DTRC;
        const float* Cp = Bp + DSC;
        __builtin_prefetch(Bp + 64, 0, 1);   // speculative; OOB dropped by HW
        float du = dt * uu;
        float y  = 0.f;
#pragma unroll
        for (int s = 0; s < DSC; ++s) {
            float hs = __expf(dt * Areg[s]) * h[s] + du * Bp[s];
            h[s] = hs;
            y += hs * Cp[s];
        }
        y += uu * Dd;
        yg[row * DIC + d] = (__bf16)(y * silu_f(zz));
    }
}

// ---------------------------------------------------------------------------
// BatchNorm over channel dim: stats (mean, rstd) then in-place apply.
// ---------------------------------------------------------------------------
__global__ __launch_bounds__(256) void bn_stats_k(const float* __restrict__ h,
                                                  float* __restrict__ stats) {
    __shared__ float s1[256], s2[256];
    int m  = blockIdx.x * 64 + (threadIdx.x & 63);
    int ty = threadIdx.x >> 6;   // 0..3
    float a = 0.f, bsum = 0.f;
    for (int r = ty; r < BNC; r += 4) {
        float v = h[(long)r * DMC + m];
        a += v;
        bsum += v * v;
    }
    s1[threadIdx.x] = a;
    s2[threadIdx.x] = bsum;
    __syncthreads();
    if (ty == 0) {
        int t = threadIdx.x;
        a    = s1[t] + s1[t + 64] + s1[t + 128] + s1[t + 192];
        bsum = s2[t] + s2[t + 64] + s2[t + 128] + s2[t + 192];
        float mean = a / (float)BNC;
        float var  = bsum / (float)BNC - mean * mean;
        stats[m]       = mean;
        stats[DMC + m] = rsqrtf(var + EPSV);
    }
}

__global__ __launch_bounds__(256) void bn_apply_k(float* __restrict__ h,
                                                  const float* __restrict__ stats,
                                                  const float* __restrict__ gamma,
                                                  const float* __restrict__ beta) {
    int idx = blockIdx.x * blockDim.x + threadIdx.x;     // BN*DM
    int m   = idx & (DMC - 1);
    h[idx]  = (h[idx] - stats[m]) * stats[DMC + m] * gamma[m] + beta[m];
}

// ---------------------------------------------------------------------------
// launch
// ---------------------------------------------------------------------------
static inline char* carve(char*& p, size_t bytes) {
    char* r = p;
    p += (bytes + 255) & ~(size_t)255;
    return r;
}

extern "C" void kernel_launch(void* const* d_in, const int* in_sizes, int n_in,
                              void* d_out, int out_size, void* d_ws, size_t ws_size,
                              hipStream_t stream) {
    const float* x        = (const float*)d_in[0];
    const float* in_w     = (const float*)d_in[1];
    const float* conv_w   = (const float*)d_in[2];
    const float* conv_b   = (const float*)d_in[3];
    const float* xproj_w  = (const float*)d_in[4];
    const float* dtproj_w = (const float*)d_in[5];
    const float* dtproj_b = (const float*)d_in[6];
    const float* A_log    = (const float*)d_in[7];
    const float* Dp       = (const float*)d_in[8];
    const float* out_w    = (const float*)d_in[9];
    const float* gamma    = (const float*)d_in[10];
    const float* beta     = (const float*)d_in[11];
    float*       out      = (float*)d_out;
    (void)in_sizes; (void)n_in; (void)out_size; (void)ws_size;

    char* p = (char*)d_ws;
    __bf16* xzbf  = (__bf16*)carve(p, (size_t)BNC * 2 * DIC * 2); // [xp|z] bf16
    float*  u     = (float*) carve(p, (size_t)BNC * DIC * 4);
    __bf16* ubf   = (__bf16*)carve(p, (size_t)BNC * DIC * 2);
    float*  dbc   = (float*) carve(p, (size_t)BNC * 64 * 4);
    __bf16* dbcbf = (__bf16*)carve(p, (size_t)BNC * 64 * 2);
    float*  dtf   = (float*) carve(p, (size_t)BNC * DIC * 4);
    __bf16* ygbf  = (__bf16*)carve(p, (size_t)BNC * DIC * 2);
    __bf16* xbf   = (__bf16*)carve(p, (size_t)BNC * DMC * 2);     // layer input (bf16)
    __bf16* inwb  = (__bf16*)carve(p, (size_t)LNUM * 2 * DIC * DMC * 2);
    __bf16* xpwb  = (__bf16*)carve(p, (size_t)LNUM * 64 * DIC * 2);
    __bf16* dtwb  = (__bf16*)carve(p, (size_t)LNUM * DIC * DTRC * 2);
    __bf16* owb   = (__bf16*)carve(p, (size_t)LNUM * DMC * DIC * 2);
    float*  stats = (float*) carve(p, (size_t)DMC * 2 * 4);

    // one-time bf16 conversions (weights + input x)
    cvt_bf16_k<<<((size_t)BNC * DMC) / 1024, 256, 0, stream>>>(x, xbf);
    cvt_bf16_k<<<((size_t)LNUM * 2 * DIC * DMC) / 1024, 256, 0, stream>>>(in_w, inwb);
    cvt_bf16_k<<<((size_t)LNUM * 64 * DIC) / 1024, 256, 0, stream>>>(xproj_w, xpwb);
    cvt_bf16_k<<<((size_t)LNUM * DIC * DTRC) / 1024, 256, 0, stream>>>(dtproj_w, dtwb);
    cvt_bf16_k<<<((size_t)LNUM * DMC * DIC) / 1024, 256, 0, stream>>>(out_w, owb);

    for (int l = 0; l < LNUM; ++l) {
        {   // in-projection (K = DM = 512) -> xzbf [BN, 2*DI] (bf16 only)
            int ntN = (2 * DIC) / 64, waves = (BNC / 16) * ntN;
            gemm_wmma_bf16<DMC, false, false, true><<<waves / 4, 128, 0, stream>>>(
                xbf, DMC, inwb + (size_t)l * 2 * DIC * DMC, nullptr,
                nullptr, xzbf, 2 * DIC, ntN);
        }
        conv_silu_k<<<(BNC * DIC) / 256, 256, 0, stream>>>(
            xzbf, conv_w + (size_t)l * DIC * DCV, conv_b + (size_t)l * DIC, u, ubf);
        {   // x-projection (K = DI = 1024) -> dbc f32 (scan) + bf16 (dt-proj A)
            int ntN = 64 / 64, waves = (BNC / 16) * ntN;
            gemm_wmma_bf16<DIC, false, true, true><<<waves / 4, 128, 0, stream>>>(
                ubf, DIC, xpwb + (size_t)l * 64 * DIC, nullptr,
                dbc, dbcbf, 64, ntN);
        }
        {   // dt-projection (K = DTR = 32) + bias + softplus -> dtf f32
            int ntN = DIC / 64, waves = (BNC / 16) * ntN;
            gemm_wmma_bf16<DTRC, true, true, false><<<waves / 4, 128, 0, stream>>>(
                dbcbf, 64, dtwb + (size_t)l * DIC * DTRC,
                dtproj_b + (size_t)l * DIC, dtf, nullptr, DIC, ntN);
        }
        scan_k<<<(BBC * DIC) / 256, 256, 0, stream>>>(
            dtf, u, xzbf, dbc, A_log + (size_t)l * DIC * DSC, Dp + (size_t)l * DIC, ygbf);
        {   // out-projection (K = DI = 1024): layer0 -> xbf (bf16), layer1 -> out (f32)
            int ntN = DMC / 64, waves = (BNC / 16) * ntN;
            if (l == LNUM - 1)
                gemm_wmma_bf16<DIC, false, true, false><<<waves / 4, 128, 0, stream>>>(
                    ygbf, DIC, owb + (size_t)l * DMC * DIC, nullptr,
                    out, nullptr, DMC, ntN);
            else
                gemm_wmma_bf16<DIC, false, false, true><<<waves / 4, 128, 0, stream>>>(
                    ygbf, DIC, owb + (size_t)l * DMC * DIC, nullptr,
                    nullptr, xbf, DMC, ntN);
        }
    }
    bn_stats_k<<<DMC / 64, 256, 0, stream>>>(out, stats);
    bn_apply_k<<<(BNC * DMC) / 256, 256, 0, stream>>>(out, stats, gamma, beta);
}
